// ModulatedConv1d_14319420965357
// MI455X (gfx1250) — matrix operand order repaired and checked by reading, not compiled, not run on hardware
//
#include <hip/hip_runtime.h>

#define BATCH 16
#define CIN   512
#define COUT  512
#define KS    3
#define NSTYLE 512
#define SIZE  4096

#define MT 64    // co rows per workgroup
#define NT 128   // i cols per workgroup
#define KT 32    // reduction step
#define LDA 40   // LDS row stride (bf16 units): 80B rows, 16B aligned, stride-20-dword bank walk
#define LDB 40

typedef __attribute__((ext_vector_type(2)))  __bf16 v2bf;
typedef __attribute__((ext_vector_type(8)))  __bf16 v8bf;
typedef __attribute__((ext_vector_type(16))) __bf16 v16bf;
typedef __attribute__((ext_vector_type(8)))  float  v8f;

union V16 { v16bf v; v8bf h[2]; };

// ---------------- kernel 1: s = leaky_relu(style @ (mod_w/sqrt(512))^T) ----------------
__global__ void k_style(const float* __restrict__ style, const float* __restrict__ modw,
                        float* __restrict__ s) {
    int idx = blockIdx.x * blockDim.x + threadIdx.x;   // b*CIN + ci
    int b = idx / CIN, ci = idx % CIN;
    const float scale = 0.044194173824159216f;         // 1/sqrt(512)
    const float* st = style + b * NSTYLE;
    const float* mw = modw + ci * NSTYLE;
    float acc = 0.f;
#pragma unroll 4
    for (int j = 0; j < NSTYLE; ++j) acc += st[j] * mw[j];
    acc *= scale;
    s[idx] = acc > 0.f ? acc : 0.01f * acc;
}

// ---------------- kernel 2: demod[b,co] = rsqrt(cs^2 * sum(weight^2 * s^2) + eps) -------
__global__ void k_demod(const float* __restrict__ w, const float* __restrict__ s,
                        float* __restrict__ demod) {
    int idx = blockIdx.x * blockDim.x + threadIdx.x;   // b*COUT + co
    int b = idx / COUT, co = idx % COUT;
    const float* wr = w + (size_t)co * CIN * KS;
    const float* sb = s + b * CIN;
    float acc = 0.f;
    for (int ci = 0; ci < CIN; ++ci) {
        float w0 = wr[ci * 3], w1 = wr[ci * 3 + 1], w2 = wr[ci * 3 + 2];
        float sv = sb[ci];
        acc += (w0 * w0 + w1 * w1 + w2 * w2) * sv * sv;
    }
    const float cs2 = 2.170138888888889e-4f;           // 1/(512*9)
    demod[idx] = rsqrtf(cs2 * acc + 1e-8f);
}

// ---------------- kernel 3: wbuf[b][k][co][ci] = bf16(cs * w * s * demod) ---------------
__global__ void k_wmod(const float* __restrict__ w, const float* __restrict__ s,
                       const float* __restrict__ demod, __bf16* __restrict__ wbuf) {
    size_t idx = (size_t)blockIdx.x * blockDim.x + threadIdx.x;
    int ci = idx % CIN; size_t t = idx / CIN;
    int co = t % COUT; t /= COUT;
    int k = t % KS;    int b = t / KS;
    const float cs = 0.014731391274719738f;            // 1/sqrt(512*9)
    float val = cs * w[((size_t)co * CIN + ci) * KS + k] * s[b * CIN + ci] * demod[b * COUT + co];
    wbuf[idx] = (__bf16)val;
}

// ---------------- kernel 4: per-batch GEMM Out[co,i] = sum_k sum_ci Wk * x_shift --------
__global__ __launch_bounds__(256) void k_conv(const float* __restrict__ x,
                                              const __bf16* __restrict__ wbuf,
                                              float* __restrict__ out) {
    __shared__ __align__(16) __bf16 Ab[MT * LDA];      // Ab[row(co)][ci_local], padded rows
    __shared__ __align__(16) __bf16 Bb[NT * LDB];      // Bb[i_local][ci_local]  (transposed)

    const int b   = blockIdx.z;
    const int co0 = blockIdx.y * MT;
    const int i0  = blockIdx.x * NT;
    const int t    = threadIdx.x;
    const int wave = t >> 5, lane = t & 31;
    const int wm = wave >> 2, wn = wave & 3;           // 2 x 4 wave grid -> 64 x 128 tile
    const int lh = lane >> 4, lm = lane & 15;

    v8f acc[2][2] = {};

    const int arow = t >> 2, aseg = t & 3;             // A staging: 64 rows x 4 16B segs
    const int bi = t & 127, bhalf = t >> 7;            // B staging: i index, ci half

    const float* xb = x + (size_t)b * CIN * SIZE;

    // LDS byte address for async A staging (generic ptr low 32 bits = LDS offset)
    const unsigned ldsA = (unsigned)(uintptr_t)(&Ab[arow * LDA + aseg * 8]);

    for (int k = 0; k < KS; ++k) {
        const __bf16* wk = wbuf + (((size_t)b * KS + k) * COUT + co0) * CIN;
        const int gi = i0 - 1 + k + bi;                // pad = 1 shift
        const bool inb = (unsigned)gi < (unsigned)SIZE;
        const size_t giC = inb ? (size_t)gi : 0;       // clamped index: load is unconditional
        const float msk = inb ? 1.f : 0.f;             // zero contribution when OOB

        for (int cb = 0; cb < CIN; cb += KT) {
            __syncthreads();
            // stage A tile: 64 x 32 bf16, async global->LDS, 16B per thread
            {
                unsigned long long gA =
                    (unsigned long long)(uintptr_t)&wk[(size_t)arow * CIN + cb + aseg * 8];
                asm volatile("global_load_async_to_lds_b128 %0, %1, off"
                             :: "v"(ldsA), "v"(gA) : "memory");
            }
            // stage B tile transposed: Bb[i][ci]; pack ci pairs -> one b32 store
#pragma unroll
            for (int rp = 0; rp < 8; ++rp) {
                int ciL = bhalf * 16 + rp * 2;
                const float* p = &xb[(size_t)(cb + ciL) * SIZE + giC];
                float v0 = p[0] * msk;
                float v1 = p[SIZE] * msk;
                v2bf pk; pk.x = (__bf16)v0; pk.y = (__bf16)v1;
                *(v2bf*)&Bb[bi * LDB + ciL] = pk;
            }
            asm volatile("s_wait_asynccnt 0x0" ::: "memory");
            __syncthreads();

            // A fragments: lane = row (lane%16); half h: K runs [8h,8h+8) and [16+8h,16+8h+8)
            V16 afr[2], bfr[2];
#pragma unroll
            for (int mi = 0; mi < 2; ++mi) {
                const __bf16* p = &Ab[(wm * 32 + mi * 16 + lm) * LDA + lh * 8];
                afr[mi].h[0] = *(const v8bf*)p;
                afr[mi].h[1] = *(const v8bf*)(p + 16);
            }
            // B fragments: lane = col (lane%16); half h: K runs [8h,8h+8) and [16+8h,16+8h+8)
#pragma unroll
            for (int ni = 0; ni < 2; ++ni) {
                const __bf16* p = &Bb[(wn * 32 + ni * 16 + lm) * LDB + lh * 8];
                bfr[ni].h[0] = *(const v8bf*)p;
                bfr[ni].h[1] = *(const v8bf*)(p + 16);
            }
#pragma unroll
            for (int mi = 0; mi < 2; ++mi)
#pragma unroll
                for (int ni = 0; ni < 2; ++ni)
                    acc[mi][ni] = __builtin_amdgcn_wmma_f32_16x16x32_bf16(
                        false, afr[mi].v, false, bfr[ni].v,
                        (short)0, acc[mi][ni], false, false);
        }
    }

    // store: C/D layout: VGPR j, lane -> M = 8*(lane/16)+j, N = lane%16
    float* ob = out + ((size_t)b * COUT + co0) * SIZE + i0;
#pragma unroll
    for (int mi = 0; mi < 2; ++mi)
#pragma unroll
        for (int ni = 0; ni < 2; ++ni) {
            int coB = wm * 32 + mi * 16 + lh * 8;
            int iB  = wn * 32 + ni * 16 + lm;
#pragma unroll
            for (int j = 0; j < 8; ++j)
                ob[(size_t)(coB + j) * SIZE + iB] = acc[mi][ni][j];
        }
}

// ---------------------------------------------------------------------------------------
extern "C" void kernel_launch(void* const* d_in, const int* in_sizes, int n_in,
                              void* d_out, int out_size, void* d_ws, size_t ws_size,
                              hipStream_t stream) {
    const float* x      = (const float*)d_in[0];   // [16,512,4096]
    const float* style  = (const float*)d_in[1];   // [16,512]
    const float* weight = (const float*)d_in[2];   // [1,512,512,3]
    const float* modw   = (const float*)d_in[3];   // [512,512]
    float* out = (float*)d_out;

    char* ws = (char*)d_ws;
    float*  s_buf  = (float*)ws;                         // 16*512 f32  (32KB)
    float*  dm_buf = (float*)(ws + 64 * 1024);           // 16*512 f32
    __bf16* wbuf   = (__bf16*)(ws + 128 * 1024);         // 16*3*512*512 bf16 (~25MB)

    k_style<<<(BATCH * CIN) / 256, 256, 0, stream>>>(style, modw, s_buf);
    k_demod<<<(BATCH * COUT) / 256, 256, 0, stream>>>(weight, s_buf, dm_buf);
    k_wmod<<<(BATCH * KS * COUT * CIN) / 256, 256, 0, stream>>>(weight, s_buf, dm_buf, wbuf);

    dim3 grid(SIZE / NT, COUT / MT, BATCH);              // batch slowest -> L2 locality
    k_conv<<<grid, 256, 0, stream>>>(x, wbuf, out);
    (void)in_sizes; (void)n_in; (void)out_size; (void)ws_size;
}
